// LLE_22376779612673
// MI455X (gfx1250) — matrix operand order repaired and checked by reading, not compiled, not run on hardware
//
#include <hip/hip_runtime.h>
#include <stdint.h>

// ---------------- Problem constants (from reference) ----------------
#define NPT   4096      // total points (2048 support + 2048 query)
#define SN    2048
#define DIM   512
#define KNN   32
#define PITERS 128      // power-iteration count

typedef __attribute__((ext_vector_type(16))) __bf16 v16bf;
typedef __attribute__((ext_vector_type(8)))  __bf16 v8bf;
typedef __attribute__((ext_vector_type(8)))  float  v8f;
typedef unsigned int u32x4 __attribute__((ext_vector_type(4)));
typedef int          i32x4 __attribute__((ext_vector_type(4)));
typedef int          i32x8 __attribute__((ext_vector_type(8)));

// ============================================================================
// TDM helper: async-load a [rows=128 x cols=32] bf16 tile from a row-major
// matrix with 4096-element row stride into LDS. Descriptor per ISA 08 §8.
// ============================================================================
__device__ __attribute__((always_inline)) inline
void tdm_load_tile_128x32(uint32_t lds_addr, uint64_t gaddr) {
    u32x4 g0;
    g0.x = 1u;                                            // count=1, user D#
    g0.y = lds_addr;                                      // LDS byte address
    g0.z = (uint32_t)gaddr;                               // global_addr[31:0]
    g0.w = ((uint32_t)(gaddr >> 32) & 0x1FFFFFFu) | (2u << 30); // addr[56:32] | type=2
    i32x8 g1;
    g1[0] = 1 << 16;                                      // data_size=1 (2 bytes)
    g1[1] = (int)(4096u << 16);                           // tensor_dim0[15:0] in [31:16]
    g1[2] = (int)(4096u << 16);                           // dim0 hi16=0, tensor_dim1 lo16 in [31:16]
    g1[3] = 32 << 16;                                     // dim1 hi16=0, tile_dim0=32 in [31:16]
    g1[4] = 128;                                          // tile_dim1=128, tile_dim2=0
    g1[5] = 4096;                                         // tensor_dim0_stride lo32
    g1[6] = 0;                                            // stride hi16, dim1_stride lo16
    g1[7] = 0;
    i32x4 z4 = {0, 0, 0, 0};
#if defined(__clang_major__) && (__clang_major__ >= 23)
    i32x8 z8 = {0, 0, 0, 0, 0, 0, 0, 0};
    __builtin_amdgcn_tensor_load_to_lds(g0, g1, z4, z4, z8, 0);
#else
    __builtin_amdgcn_tensor_load_to_lds(g0, g1, z4, z4, 0);
#endif
}

// ============================================================================
// K1: concat support+query -> fp32 master, bf16 shadow, row sq-norms
// ============================================================================
__global__ void lle_pack(const float* __restrict__ sup, const float* __restrict__ qry,
                         float* __restrict__ dataf, __bf16* __restrict__ databf,
                         float* __restrict__ sq) {
    int i = blockIdx.x;          // row 0..4095
    int t = threadIdx.x;         // 256 threads
    const float* src = (i < SN) ? (sup + (size_t)i * DIM) : (qry + (size_t)(i - SN) * DIM);
    float s = 0.f;
    for (int d = t; d < DIM; d += 256) {
        float v = src[d];
        dataf[(size_t)i * DIM + d]  = v;
        databf[(size_t)i * DIM + d] = (__bf16)v;
        s += v * v;
    }
    __shared__ float red[256];
    red[t] = s; __syncthreads();
    for (int o = 128; o > 0; o >>= 1) { if (t < o) red[t] += red[t + o]; __syncthreads(); }
    if (t == 0) sq[i] = red[0];
}

// ============================================================================
// K2: pairwise squared distances via WMMA Gram (one wave per 16x16 tile)
//     dist[i][j] = sq[i] + sq[j] - 2 * <data_i, data_j>
// ============================================================================
__global__ void __launch_bounds__(32) lle_dist(const __bf16* __restrict__ db,
                                               const float* __restrict__ sq,
                                               float* __restrict__ dist) {
    const int ti = blockIdx.y, tj = blockIdx.x;
    const int lane = threadIdx.x, half = lane >> 4, l = lane & 15;
    const __bf16* Ar = db + (size_t)(ti * 16 + l) * DIM;
    const __bf16* Br = db + (size_t)(tj * 16 + l) * DIM;
    v8f acc = {};
    for (int k0 = 0; k0 < DIM; k0 += 32) {
        __builtin_prefetch(Ar + k0 + 128, 0, 0);
        __builtin_prefetch(Br + k0 + 128, 0, 0);
        v8bf a0 = *(const v8bf*)(Ar + k0 + half * 8);
        v8bf a1 = *(const v8bf*)(Ar + k0 + 16 + half * 8);
        v16bf av;
        #pragma unroll
        for (int e = 0; e < 8; ++e) { av[e] = a0[e]; av[e + 8] = a1[e]; }
        v8bf b0 = *(const v8bf*)(Br + k0 + half * 8);
        v8bf b1 = *(const v8bf*)(Br + k0 + 16 + half * 8);
        v16bf bv;
        #pragma unroll
        for (int e = 0; e < 8; ++e) { bv[e] = b0[e]; bv[e + 8] = b1[e]; }
        acc = __builtin_amdgcn_wmma_f32_16x16x32_bf16(false, av, false, bv,
                                                      (short)0, acc, false, false);
    }
    const float sqc = sq[tj * 16 + l];
    #pragma unroll
    for (int r = 0; r < 8; ++r) {
        int row = ti * 16 + half * 8 + r;
        dist[(size_t)row * NPT + tj * 16 + l] = sq[row] + sqc - 2.0f * acc[r];
    }
}

// ============================================================================
// K3: per-row 32-NN selection (exclude self), lexicographic repeated argmin
// ============================================================================
__global__ void lle_select(const float* __restrict__ dist, int* __restrict__ idx) {
    int row = blockIdx.x, t = threadIdx.x;
    const float* dr = dist + (size_t)row * NPT;
    __shared__ float sv[256]; __shared__ int si[256];
    float lastV = -3.0e38f; int lastI = -1;
    for (int s = 0; s < KNN; ++s) {
        float bv = 3.0e38f; int bi = 0x7fffffff;
        for (int j = t; j < NPT; j += 256) {
            if (j == row) continue;
            float v = dr[j];
            bool gtLast = (v > lastV) || (v == lastV && j > lastI);
            if (gtLast && (v < bv || (v == bv && j < bi))) { bv = v; bi = j; }
        }
        sv[t] = bv; si[t] = bi; __syncthreads();
        for (int o = 128; o > 0; o >>= 1) {
            if (t < o) {
                if (sv[t + o] < sv[t] || (sv[t + o] == sv[t] && si[t + o] < si[t])) {
                    sv[t] = sv[t + o]; si[t] = si[t + o];
                }
            }
            __syncthreads();
        }
        if (t == 0) idx[(size_t)row * KNN + s] = si[0];
        lastV = sv[0]; lastI = si[0];
        __syncthreads();
    }
}

// ============================================================================
// K4: barycenter weights. One block (128 thr = 4 waves) per point.
//     Residuals staged bf16 in LDS; G = res res^T via WMMA; GJ solve on 1 wave.
// ============================================================================
__global__ void __launch_bounds__(128) lle_bary(const float* __restrict__ dataf,
                                                const int* __restrict__ idx,
                                                float* __restrict__ wts) {
    __shared__ __align__(32) __bf16 res[KNN][DIM + 16];
    __shared__ float G[KNN][36];
    __shared__ float piv[33];
    __shared__ float wv[KNN];
    int i = blockIdx.x, t = threadIdx.x;
    const int* ip = idx + (size_t)i * KNN;
    const float* ctr = dataf + (size_t)i * DIM;
    for (int r = 0; r < KNN; ++r) {
        const float* src = dataf + (size_t)ip[r] * DIM;
        for (int d = t; d < DIM; d += 128)
            res[r][d] = (__bf16)(src[d] - ctr[d]);
    }
    __syncthreads();

    {
        int w = t >> 5, lane = t & 31, half = lane >> 4, l = lane & 15;
        int tm = (w >> 1) * 16, tn = (w & 1) * 16;
        const __bf16* Ar = &res[tm + l][0];
        const __bf16* Br = &res[tn + l][0];
        v8f acc = {};
        for (int k0 = 0; k0 < DIM; k0 += 32) {
            v8bf a0 = *(const v8bf*)(Ar + k0 + half * 8);
            v8bf a1 = *(const v8bf*)(Ar + k0 + 16 + half * 8);
            v16bf av;
            #pragma unroll
            for (int e = 0; e < 8; ++e) { av[e] = a0[e]; av[e + 8] = a1[e]; }
            v8bf b0 = *(const v8bf*)(Br + k0 + half * 8);
            v8bf b1 = *(const v8bf*)(Br + k0 + 16 + half * 8);
            v16bf bv;
            #pragma unroll
            for (int e = 0; e < 8; ++e) { bv[e] = b0[e]; bv[e + 8] = b1[e]; }
            acc = __builtin_amdgcn_wmma_f32_16x16x32_bf16(false, av, false, bv,
                                                          (short)0, acc, false, false);
        }
        #pragma unroll
        for (int r = 0; r < 8; ++r)
            G[tm + half * 8 + r][tn + l] = acc[r];
    }
    __syncthreads();

    float Ar[33];
    if (t < KNN) {
        float tr = 0.f;
        for (int d = 0; d < KNN; ++d) tr += G[d][d];
        float R = (tr > 0.f) ? 1e-3f * tr : 1e-3f;
        for (int c = 0; c < KNN; ++c) Ar[c] = G[t][c];
        Ar[t] += R;
        Ar[32] = 1.f;
    }
    for (int c = 0; c < KNN; ++c) {
        if (t == c) { for (int j = 0; j < 33; ++j) piv[j] = Ar[j]; }
        __syncthreads();
        if (t < KNN && t != c) {
            float f = Ar[c] / piv[c];
            for (int j = c; j < 33; ++j) Ar[j] -= f * piv[j];
        }
        __syncthreads();
    }
    if (t < KNN) wv[t] = Ar[32] / Ar[t];
    __syncthreads();
    if (t < KNN) {
        float s = 0.f;
        for (int j = 0; j < KNN; ++j) s += wv[j];
        wts[(size_t)i * KNN + t] = wv[t] / s;
    }
}

// ============================================================================
// K5: build IWT = (I - W)^T in bf16 (identity init + weight scatter)
// ============================================================================
__global__ void lle_iwt_init(__bf16* __restrict__ iwt) {
    size_t p = (size_t)blockIdx.x * 256 + threadIdx.x;
    int i = (int)(p >> 12), j = (int)(p & (NPT - 1));
    iwt[p] = (i == j) ? (__bf16)1.0f : (__bf16)0.0f;
}
__global__ void lle_iwt_scatter(const int* __restrict__ idx, const float* __restrict__ wts,
                                __bf16* __restrict__ iwt) {
    int p = blockIdx.x * 256 + threadIdx.x;
    int j = p >> 5;
    int r = idx[p];
    iwt[(size_t)r * NPT + j] = (__bf16)(-wts[p]);
}

// ============================================================================
// K6: M = IWT * IWT^T via LDS-tiled WMMA GEMM with TDM double buffering.
//     Block = 256 thr (8 waves, 2x4), block tile 128x128, K-step 32.
//     Wave 0 drives the Tensor Data Mover; s_wait_tensorcnt + barrier publish.
// ============================================================================
__global__ void __launch_bounds__(256) lle_mgemm(const __bf16* __restrict__ iwt,
                                                 float* __restrict__ Mout) {
    __shared__ __align__(16) __bf16 As[2][128][32];
    __shared__ __align__(16) __bf16 Bs[2][128][32];
    const int t = threadIdx.x;
    const int wave = t >> 5, lane = t & 31, half = lane >> 4, l = lane & 15;
    const int wr = wave >> 2, wc = wave & 3;          // 2 x 4 wave grid
    const int m0 = wr * 64, n0 = wc * 32;
    const int bi = blockIdx.y * 128, bj = blockIdx.x * 128;

    const uint64_t gA = (uint64_t)(uintptr_t)(iwt + (size_t)bi * NPT);
    const uint64_t gB = (uint64_t)(uintptr_t)(iwt + (size_t)bj * NPT);
    const uint32_t lA[2] = { (uint32_t)(uintptr_t)&As[0][0][0],
                             (uint32_t)(uintptr_t)&As[1][0][0] };
    const uint32_t lB[2] = { (uint32_t)(uintptr_t)&Bs[0][0][0],
                             (uint32_t)(uintptr_t)&Bs[1][0][0] };

    if (wave == 0) {
        tdm_load_tile_128x32(lA[0], gA);
        tdm_load_tile_128x32(lB[0], gB);
    }

    v8f acc[4][2] = {};
    for (int s = 0; s < NPT / 32; ++s) {
        const int cur = s & 1;
        if (wave == 0) {
            if (s + 1 < NPT / 32) {
                const uint64_t koff = (uint64_t)(s + 1) * 64;   // 32 bf16 = 64 bytes
                tdm_load_tile_128x32(lA[cur ^ 1], gA + koff);
                tdm_load_tile_128x32(lB[cur ^ 1], gB + koff);
                __builtin_amdgcn_s_wait_tensorcnt(2);           // current pair done
            } else {
                __builtin_amdgcn_s_wait_tensorcnt(0);
            }
        }
        __syncthreads();   // publish LDS tile to all waves

        v16bf bfr[2];
        #pragma unroll
        for (int ni = 0; ni < 2; ++ni) {
            v8bf b0 = *(const v8bf*)&Bs[cur][n0 + ni * 16 + l][half * 8];
            v8bf b1 = *(const v8bf*)&Bs[cur][n0 + ni * 16 + l][16 + half * 8];
            #pragma unroll
            for (int e = 0; e < 8; ++e) { bfr[ni][e] = b0[e]; bfr[ni][e + 8] = b1[e]; }
        }
        #pragma unroll
        for (int mi = 0; mi < 4; ++mi) {
            v8bf a0 = *(const v8bf*)&As[cur][m0 + mi * 16 + l][half * 8];
            v8bf a1 = *(const v8bf*)&As[cur][m0 + mi * 16 + l][16 + half * 8];
            v16bf av;
            #pragma unroll
            for (int e = 0; e < 8; ++e) { av[e] = a0[e]; av[e + 8] = a1[e]; }
            #pragma unroll
            for (int ni = 0; ni < 2; ++ni)
                acc[mi][ni] = __builtin_amdgcn_wmma_f32_16x16x32_bf16(
                    false, av, false, bfr[ni], (short)0, acc[mi][ni], false, false);
        }
        __syncthreads();   // all reads done before buffer is overwritten
    }

    #pragma unroll
    for (int mi = 0; mi < 4; ++mi)
        #pragma unroll
        for (int ni = 0; ni < 2; ++ni)
            #pragma unroll
            for (int r = 0; r < 8; ++r) {
                int row = bi + m0 + mi * 16 + half * 8 + r;
                int col = bj + n0 + ni * 16 + l;
                Mout[(size_t)row * NPT + col] = acc[mi][ni][r];
            }
}

// ============================================================================
// K7: eigen machinery (deflated block power iteration on sigma*I - M)
// ============================================================================
__global__ void lle_zero(float* __restrict__ s) { if (threadIdx.x < 64) s[threadIdx.x] = 0.f; }

__global__ void lle_rowsum(const float* __restrict__ Mm, float* __restrict__ rs) {
    int i = blockIdx.x, t = threadIdx.x;
    const float* r = Mm + (size_t)i * NPT;
    float s = 0.f;
    for (int j = t; j < NPT; j += 256) s += fabsf(r[j]);
    __shared__ float buf[256];
    buf[t] = s; __syncthreads();
    for (int o = 128; o > 0; o >>= 1) { if (t < o) buf[t] += buf[t + o]; __syncthreads(); }
    if (t == 0) rs[i] = buf[0];
}
__global__ void lle_maxred(const float* __restrict__ rs, float* __restrict__ scal) {
    int t = threadIdx.x;
    float m = 0.f;
    for (int i = t; i < NPT; i += 1024) m = fmaxf(m, rs[i]);
    __shared__ float buf[1024];
    buf[t] = m; __syncthreads();
    for (int o = 512; o > 0; o >>= 1) { if (t < o) buf[t] = fmaxf(buf[t], buf[t + o]); __syncthreads(); }
    if (t == 0) scal[0] = buf[0] * 1.0001f + 1e-6f;
}
__global__ void lle_initX(float* __restrict__ X) {
    int p = blockIdx.x * 256 + threadIdx.x;
    if (p >= 3 * NPT) return;
    unsigned u = (unsigned)p * 2654435761u ^ 0x9E3779B9u;
    u ^= u >> 16; u *= 0x85EBCA6Bu; u ^= u >> 13;
    X[p] = (float)(u & 0xFFFFFF) / 8388608.0f - 1.0f;
}
__global__ void lle_matvec(const float* __restrict__ Mm, const float* __restrict__ X,
                           float* __restrict__ Y) {
    int i = blockIdx.x, t = threadIdx.x;
    const float* Mr = Mm + (size_t)i * NPT;
    float a0 = 0.f, a1 = 0.f, a2 = 0.f;
    for (int j = t; j < NPT; j += 256) {
        float m = Mr[j];
        a0 += m * X[j]; a1 += m * X[NPT + j]; a2 += m * X[2 * NPT + j];
    }
    __shared__ float r0[256], r1[256], r2[256];
    r0[t] = a0; r1[t] = a1; r2[t] = a2; __syncthreads();
    for (int o = 128; o > 0; o >>= 1) {
        if (t < o) { r0[t] += r0[t + o]; r1[t] += r1[t + o]; r2[t] += r2[t + o]; }
        __syncthreads();
    }
    if (t == 0) { Y[i] = r0[0]; Y[NPT + i] = r1[0]; Y[2 * NPT + i] = r2[0]; }
}
__global__ void lle_dots(const float* __restrict__ X, const float* __restrict__ Y,
                         float* __restrict__ scal) {
    int t = threadIdx.x;
    float sg = scal[0];
    float s0 = 0, s1 = 0, s2 = 0, g0 = 0, g1 = 0, g2 = 0, g3 = 0, g4 = 0, g5 = 0;
    for (int i = t; i < NPT; i += 1024) {
        float z0 = sg * X[i]           - Y[i];
        float z1 = sg * X[NPT + i]     - Y[NPT + i];
        float z2 = sg * X[2 * NPT + i] - Y[2 * NPT + i];
        s0 += z0; s1 += z1; s2 += z2;
        g0 += z0 * z0; g1 += z0 * z1; g2 += z0 * z2;
        g3 += z1 * z1; g4 += z1 * z2; g5 += z2 * z2;
    }
    float acc[9] = { s0, s1, s2, g0, g1, g2, g3, g4, g5 };
    __shared__ float buf[1024];
    for (int q = 0; q < 9; ++q) {
        buf[t] = acc[q]; __syncthreads();
        for (int o = 512; o > 0; o >>= 1) { if (t < o) buf[t] += buf[t + o]; __syncthreads(); }
        if (t == 0) scal[1 + q] = buf[0];
        __syncthreads();
    }
}
__global__ void lle_coeff(float* __restrict__ scal) {
    if (threadIdx.x != 0 || blockIdx.x != 0) return;
    const float n = (float)NPT;
    float s0 = scal[1], s1 = scal[2], s2 = scal[3];
    float g00 = scal[4] - s0 * s0 / n, g01 = scal[5] - s0 * s1 / n, g02 = scal[6] - s0 * s2 / n;
    float g11 = scal[7] - s1 * s1 / n, g12 = scal[8] - s1 * s2 / n, g22 = scal[9] - s2 * s2 / n;
    float eps = 1e-20f + 1e-12f * (g00 + g11 + g22);
    float l00 = sqrtf(fmaxf(g00, eps));
    float l10 = g01 / l00, l20 = g02 / l00;
    float l11 = sqrtf(fmaxf(g11 - l10 * l10, eps));
    float l21 = (g12 - l20 * l10) / l11;
    float l22 = sqrtf(fmaxf(g22 - l20 * l20 - l21 * l21, eps));
    float i00 = 1.f / l00, i11 = 1.f / l11, i22 = 1.f / l22;
    float i10 = -l10 * i00 * i11;
    float i21 = -l21 * i11 * i22;
    float i20 = (l10 * l21 - l20 * l11) * i00 * i11 * i22;
    float M3[9] = { i00, 0, 0, i10, i11, 0, i20, i21, i22 };
    for (int q = 0; q < 9; ++q) scal[10 + q] = M3[q];
    float m0 = s0 / n, m1 = s1 / n, m2 = s2 / n;
    scal[19] = -(i00 * m0);
    scal[20] = -(i10 * m0 + i11 * m1);
    scal[21] = -(i20 * m0 + i21 * m1 + i22 * m2);
}
__global__ void lle_update(float* __restrict__ X, const float* __restrict__ Y,
                           const float* __restrict__ scal) {
    int i = blockIdx.x * 256 + threadIdx.x;
    if (i >= NPT) return;
    float sg = scal[0];
    float z0 = sg * X[i]           - Y[i];
    float z1 = sg * X[NPT + i]     - Y[NPT + i];
    float z2 = sg * X[2 * NPT + i] - Y[2 * NPT + i];
    float n0 = scal[10] * z0 + scal[11] * z1 + scal[12] * z2 + scal[19];
    float n1 = scal[13] * z0 + scal[14] * z1 + scal[15] * z2 + scal[20];
    float n2 = scal[16] * z0 + scal[17] * z1 + scal[18] * z2 + scal[21];
    X[i] = n0; X[NPT + i] = n1; X[2 * NPT + i] = n2;
}
__global__ void lle_ray(const float* __restrict__ X, const float* __restrict__ Y,
                        float* __restrict__ scal) {
    int t = threadIdx.x;
    float xy[3] = {0, 0, 0}, xx[3] = {0, 0, 0};
    for (int i = t; i < NPT; i += 1024) {
        #pragma unroll
        for (int a = 0; a < 3; ++a) {
            float x = X[a * NPT + i];
            xy[a] += x * Y[a * NPT + i];
            xx[a] += x * x;
        }
    }
    __shared__ float buf[1024];
    for (int a = 0; a < 3; ++a) {
        buf[t] = xy[a]; __syncthreads();
        for (int o = 512; o > 0; o >>= 1) { if (t < o) buf[t] += buf[t + o]; __syncthreads(); }
        float num = buf[0]; __syncthreads();
        buf[t] = xx[a]; __syncthreads();
        for (int o = 512; o > 0; o >>= 1) { if (t < o) buf[t] += buf[t + o]; __syncthreads(); }
        if (t == 0) scal[22 + a] = num / fmaxf(buf[0], 1e-30f);
        __syncthreads();
    }
}
__global__ void lle_out(const float* __restrict__ X, const float* __restrict__ scal,
                        float* __restrict__ out) {
    int i = blockIdx.x * 256 + threadIdx.x;
    if (i >= NPT) return;
    float r[3] = { scal[22], scal[23], scal[24] };
    int ord[3] = { 0, 1, 2 };
    if (r[ord[1]] < r[ord[0]]) { int x = ord[0]; ord[0] = ord[1]; ord[1] = x; }
    if (r[ord[2]] < r[ord[1]]) { int x = ord[1]; ord[1] = ord[2]; ord[2] = x; }
    if (r[ord[1]] < r[ord[0]]) { int x = ord[0]; ord[0] = ord[1]; ord[1] = x; }
    out[2 * i + 0] = X[ord[0] * NPT + i];
    out[2 * i + 1] = X[ord[1] * NPT + i];
}

// ============================================================================
// Host orchestration
// ============================================================================
extern "C" void kernel_launch(void* const* d_in, const int* in_sizes, int n_in,
                              void* d_out, int out_size, void* d_ws, size_t ws_size,
                              hipStream_t stream) {
    (void)in_sizes; (void)n_in; (void)out_size; (void)ws_size;
    const float* sup = (const float*)d_in[0];
    const float* qry = (const float*)d_in[1];
    float* out = (float*)d_out;

    char* w = (char*)d_ws;
    size_t off = 0;
    auto place = [&](size_t bytes) -> char* {
        char* p = w + off;
        off = (off + bytes + 255) & ~(size_t)255;
        return p;
    };
    float*  dataf  = (float*) place((size_t)NPT * DIM * 4);
    __bf16* databf = (__bf16*)place((size_t)NPT * DIM * 2);
    float*  sq     = (float*) place((size_t)NPT * 4);
    float*  dist   = (float*) place((size_t)NPT * NPT * 4);   // reused as M
    int*    idxb   = (int*)   place((size_t)NPT * KNN * 4);
    float*  wts    = (float*) place((size_t)NPT * KNN * 4);
    __bf16* iwt    = (__bf16*)place((size_t)NPT * NPT * 2);
    float*  X      = (float*) place((size_t)3 * NPT * 4);
    float*  Y      = (float*) place((size_t)3 * NPT * 4);
    float*  scal   = (float*) place(256);
    float*  rs     = (float*) place((size_t)NPT * 4);
    float*  Mm     = dist;

    lle_zero<<<1, 64, 0, stream>>>(scal);
    lle_pack<<<NPT, 256, 0, stream>>>(sup, qry, dataf, databf, sq);
    lle_dist<<<dim3(NPT / 16, NPT / 16), 32, 0, stream>>>(databf, sq, dist);
    lle_select<<<NPT, 256, 0, stream>>>(dist, idxb);
    lle_bary<<<NPT, 128, 0, stream>>>(dataf, idxb, wts);
    lle_iwt_init<<<(NPT * (NPT / 256)), 256, 0, stream>>>(iwt);
    lle_iwt_scatter<<<(NPT * KNN) / 256, 256, 0, stream>>>(idxb, wts, iwt);
    lle_mgemm<<<dim3(NPT / 128, NPT / 128), 256, 0, stream>>>(iwt, Mm);
    lle_rowsum<<<NPT, 256, 0, stream>>>(Mm, rs);
    lle_maxred<<<1, 1024, 0, stream>>>(rs, scal);
    lle_initX<<<(3 * NPT + 255) / 256, 256, 0, stream>>>(X);
    for (int it = 0; it < PITERS; ++it) {
        lle_matvec<<<NPT, 256, 0, stream>>>(Mm, X, Y);
        lle_dots<<<1, 1024, 0, stream>>>(X, Y, scal);
        lle_coeff<<<1, 1, 0, stream>>>(scal);
        lle_update<<<NPT / 256, 256, 0, stream>>>(X, Y, scal);
    }
    lle_matvec<<<NPT, 256, 0, stream>>>(Mm, X, Y);
    lle_ray<<<1, 1024, 0, stream>>>(X, Y, scal);
    lle_out<<<NPT / 256, 256, 0, stream>>>(X, scal, out);
}
